// IAttention_19533511262641
// MI455X (gfx1250) — matrix operand order repaired and checked by reading, not compiled, not run on hardware
//
#include <hip/hip_runtime.h>
#include <hip/hip_bf16.h>
#include <math.h>

// ---------------------------------------------------------------------------
// Types / helpers
// ---------------------------------------------------------------------------
typedef __attribute__((ext_vector_type(16))) __bf16 v16bf;
typedef __attribute__((ext_vector_type(8)))  __bf16 v8bf;
typedef __attribute__((ext_vector_type(8)))  float  v8f;
typedef unsigned short u16;                         // raw bf16 bits
typedef __attribute__((address_space(3))) u16 u16_lds;

union ABFrag { v16bf v; v8bf h[2]; };

__device__ __forceinline__ u16 f2bf(float f) {
  union { float f; unsigned u; } c; c.f = f;
  unsigned u = c.u;
  unsigned r = u + 0x7FFFu + ((u >> 16) & 1u); // round-to-nearest-even
  return (u16)(r >> 16);
}

// A-fragment (16x32 bf16, M x K). Per ISA: lane<16 row M=lane holds K {0..7,16..23},
// lane>=16 row M=lane-16 holds K {8..15,24..31}. Two contiguous 16B loads.
__device__ __forceinline__ v16bf load_a_frag(const u16* A, int lda, int m0, int k0) {
  int lane = threadIdx.x & 31;
  int hf   = lane >> 4;
  int r    = lane & 15;
  const u16* p = A + (size_t)(m0 + r) * lda + k0 + hf * 8;
  ABFrag f;
  f.h[0] = *reinterpret_cast<const v8bf*>(p);
  f.h[1] = *reinterpret_cast<const v8bf*>(p + 16);
  return f.v;
}

// B-fragment (32x16 bf16, K x N) from an [N,K] buffer.
// lane<16: column N=n0+lane, K = k0+{0..7,16..23}; lane>=16: K = k0+{8..15,24..31}.
__device__ __forceinline__ v16bf load_b_frag(const u16* Bt, int ldb, int n0, int k0) {
  int lane = threadIdx.x & 31;
  int hf   = lane >> 4;
  int c    = lane & 15;
  const u16* p = Bt + (size_t)(n0 + c) * ldb + k0 + hf * 8;
  ABFrag f;
  f.h[0] = *reinterpret_cast<const v8bf*>(p);
  f.h[1] = *reinterpret_cast<const v8bf*>(p + 16);
  return f.v;
}

#define WMMA_BF16(a, b, c) \
  __builtin_amdgcn_wmma_f32_16x16x32_bf16(false, (a), false, (b), (short)0, (c), false, false)

// gfx1250 async global->LDS copy (ASYNCcnt-tracked DMA into LDS).
__device__ __forceinline__ void async_load_b128(const u16* gsrc, u16* lds_dst) {
  u16_lds* lp = (u16_lds*)lds_dst; // 32-bit LDS offset in a VGPR
  asm volatile("global_load_async_to_lds_b128 %0, %1, off"
               :: "v"(lp), "v"(gsrc) : "memory");
}

__device__ __forceinline__ void wait_async0() {
#if __has_builtin(__builtin_amdgcn_s_wait_asynccnt)
  __builtin_amdgcn_s_wait_asynccnt(0);
#else
  asm volatile("s_wait_asynccnt 0x0" ::: "memory");
#endif
}

// ---------------------------------------------------------------------------
// Elementwise convert / repack kernels
// ---------------------------------------------------------------------------
__global__ void k_f32_to_bf16(const float* __restrict__ in, u16* __restrict__ out, size_t n) {
  size_t i = (size_t)blockIdx.x * blockDim.x + threadIdx.x;
  if (i < n) out[i] = f2bf(in[i]);
}

// additive key bias: mask==0 -> -1e9, else 0
__global__ void k_mask_bias(const int* __restrict__ mask, float* __restrict__ out, size_t n) {
  size_t i = (size_t)blockIdx.x * blockDim.x + threadIdx.x;
  if (i < n) out[i] = mask[i] ? 0.f : -1.0e9f;
}

// Wq [H,E,D] -> out[(h*D+d)*E + e]  (N-major, K=E contiguous)
__global__ void k_repack_qkv(const float* __restrict__ in, u16* __restrict__ out,
                             int E, int D, size_t n) {
  size_t o = (size_t)blockIdx.x * blockDim.x + threadIdx.x;
  if (o >= n) return;
  int e = (int)(o % E);
  size_t nd = o / E;
  int d = (int)(nd % D);
  int h = (int)(nd / D);
  out[o] = f2bf(in[((size_t)h * E + e) * D + d]);
}

// in [K,N] f32 -> out [N,K] bf16
__global__ void k_transpose_bf16(const float* __restrict__ in, u16* __restrict__ out,
                                 int K, int N, size_t n) {
  size_t o = (size_t)blockIdx.x * blockDim.x + threadIdx.x;
  if (o >= n) return;
  int k = (int)(o % K);
  int nn = (int)(o / K);
  out[o] = f2bf(in[(size_t)k * N + nn]);
}

// ---------------------------------------------------------------------------
// Generic WMMA GEMM:  Y[T,N] = A[T,K](bf16) * Bt[N,K](bf16) + bias[N]
// 8 waves / block, one 32x32 C tile per wave.
// B panel (shared by all 8 waves) staged in LDS via async global->LDS DMA,
// double-buffered in 64-wide K stages: each of the 256 threads issues exactly
// one global_load_async_to_lds_b128 per stage (4 KB panel); 8 WMMAs per sync.
// MODE: 0 = scatter bf16 to [B,H,S,D] (Q,K)     1 = scatter bf16 to [B,H,D,S] (V^T)
//       2 = plain f32 [T,N]                      3 = exact GELU -> bf16 [T,N]
// ---------------------------------------------------------------------------
template <int MODE>
__global__ void k_gemm_bf16(const u16* __restrict__ A, const u16* __restrict__ Bt,
                            const float* __restrict__ bias, void* __restrict__ out,
                            int T, int K, int N, int S, int H) {
  // [buf][ncol 32][k 64 padded to 72] : rows 144B (9x16B) -> aligned 16B frags
  __shared__ __align__(16) u16 btile[2][32][72];

  int tid  = (int)threadIdx.x;
  int wave = tid >> 5;
  int m0   = (blockIdx.y * 8 + wave) * 32;
  int n0   = blockIdx.x * 32;

  int colb = tid >> 3;      // 0..31 : B column
  int ch   = tid & 7;       // 0..7  : 8-half chunk within 64-wide K stage

  // stage first B panel (k = 0..63)
  async_load_b128(Bt + (size_t)(n0 + colb) * K + ch * 8, &btile[0][colb][ch * 8]);

  v8f acc[2][2];
#pragma unroll
  for (int mt = 0; mt < 2; ++mt)
#pragma unroll
    for (int nt = 0; nt < 2; ++nt)
      acc[mt][nt] = v8f{0.f, 0.f, 0.f, 0.f, 0.f, 0.f, 0.f, 0.f};

  wait_async0();
  __syncthreads();

  int buf = 0;
  for (int k0 = 0; k0 < K; k0 += 64) {
    if (k0 + 64 < K) // stage next 64-wide panel while computing this one
      async_load_b128(Bt + (size_t)(n0 + colb) * K + (k0 + 64) + ch * 8,
                      &btile[buf ^ 1][colb][ch * 8]);

#pragma unroll
    for (int ks = 0; ks < 64; ks += 32) {
      v16bf a0 = load_a_frag(A, K, m0,      k0 + ks);
      v16bf a1 = load_a_frag(A, K, m0 + 16, k0 + ks);
      v16bf b0 = load_b_frag(&btile[buf][0][0], 72, 0,  ks);
      v16bf b1 = load_b_frag(&btile[buf][0][0], 72, 16, ks);
      acc[0][0] = WMMA_BF16(a0, b0, acc[0][0]);
      acc[0][1] = WMMA_BF16(a0, b1, acc[0][1]);
      acc[1][0] = WMMA_BF16(a1, b0, acc[1][0]);
      acc[1][1] = WMMA_BF16(a1, b1, acc[1][1]);
    }

    if (k0 + 64 < K) {
      wait_async0();
      __syncthreads();
    }
    buf ^= 1;
  }

  int lane = tid & 31;
  int hf   = lane >> 4;
  int col  = lane & 15;

#pragma unroll
  for (int mt = 0; mt < 2; ++mt) {
#pragma unroll
    for (int nt = 0; nt < 2; ++nt) {
      int n    = n0 + nt * 16 + col;
      float bv = bias[n];
#pragma unroll
      for (int i = 0; i < 8; ++i) {
        int m   = m0 + mt * 16 + hf * 8 + i; // C: VGPR i -> row i (lanes 0-15) / 8+i
        float v = acc[mt][nt][i] + bv;
        if (MODE == 2) {
          ((float*)out)[(size_t)m * N + n] = v;
        } else if (MODE == 3) {
          float g = 0.5f * v * (1.f + erff(v * 0.70710678118654752f));
          ((u16*)out)[(size_t)m * N + n] = f2bf(g);
        } else {
          int bb = m / S, s = m % S;
          int hh = n >> 6, d = n & 63; // D = 64
          if (MODE == 0)
            ((u16*)out)[(((size_t)(bb * H + hh) * S + s) << 6) + d] = f2bf(v);
          else
            ((u16*)out)[((size_t)(bb * H + hh) * 64 + d) * S + s] = f2bf(v);
        }
      }
    }
  }
}

// ---------------------------------------------------------------------------
// Flash attention, fully TRANSPOSED dataflow: per lane owns ONE query column.
//   scores^T = K_tile(A) x Q^T(B)   -> C: M=key (VGPRs), N=query (lanes)
//   O^T      = V^T(A)   x P(B)      -> C: M=d   (VGPRs), N=query (lanes)
// Row stats (m,l) are per-lane scalars; only 2 xor-16 shuffles per 32-key chunk.
// q,k : [B,H,S,64] bf16   vT : [B,H,64,S] bf16   ctx : [B*S, H*64] bf16
// kbias: additive mask bias [B,S] f32. scale = 1/sqrt(S) (reference semantics).
// ---------------------------------------------------------------------------
__global__ void k_flash_attn(const u16* __restrict__ q, const u16* __restrict__ k,
                             const u16* __restrict__ vT, const float* __restrict__ kbias,
                             u16* __restrict__ ctx, int B, int H, int S) {
  __shared__ __align__(16) u16 pbuf[4][16][32]; // [wave][query][key]

  int bh   = blockIdx.y;
  int b    = bh / H;
  int hh   = bh % H;
  int wave = threadIdx.x >> 5;
  int lane = threadIdx.x & 31;
  int hf   = lane >> 4;
  int col  = lane & 15;
  int m0   = blockIdx.x * 64 + wave * 16; // query tile start

  const u16*   Qh = q  + (size_t)bh * S * 64;
  const u16*   Kh = k  + (size_t)bh * S * 64;
  const u16*   Vh = vT + (size_t)bh * 64 * S;
  const float* bp = kbias + (size_t)b * S;

  const float scale = rsqrtf((float)S);

  // Q fragments as B operands (contraction = d), held for the whole key sweep
  v16bf qb0 = load_b_frag(Qh, 64, m0, 0);
  v16bf qb1 = load_b_frag(Qh, 64, m0, 32);

  float rm = -1.0e30f, rl = 0.f;
  v8f acc[4]; // O^T: acc[t] covers d = t*16 + (hf*8 + i), query = col
#pragma unroll
  for (int t = 0; t < 4; ++t) acc[t] = v8f{0.f, 0.f, 0.f, 0.f, 0.f, 0.f, 0.f, 0.f};

  for (int kb = 0; kb < S; kb += 32) {
    if (kb + 32 < S) { // prefetch next K/V chunks (global_prefetch_b8)
      __builtin_prefetch((const void*)(Kh + (size_t)(kb + 32) * 64), 0, 1);
      __builtin_prefetch((const void*)(Vh + (size_t)kb + 32), 0, 1);
    }

    // ---- transposed scores for two 16-key tiles --------------------------
    v8f s0 = {0.f, 0.f, 0.f, 0.f, 0.f, 0.f, 0.f, 0.f};
    v8f s1 = s0;
    s0 = WMMA_BF16(load_a_frag(Kh, 64, kb,      0),  qb0, s0);
    s0 = WMMA_BF16(load_a_frag(Kh, 64, kb,      32), qb1, s0);
    s1 = WMMA_BF16(load_a_frag(Kh, 64, kb + 16, 0),  qb0, s1);
    s1 = WMMA_BF16(load_a_frag(Kh, 64, kb + 16, 32), qb1, s1);

    // ---- additive mask bias for this lane's 16 keys (broadcast float4s) --
    float b0[8], b1[8];
    const float* kp = bp + kb + hf * 8;
    *(float4*)&b0[0] = *(const float4*)(kp);
    *(float4*)&b0[4] = *(const float4*)(kp + 4);
    *(float4*)&b1[0] = *(const float4*)(kp + 16);
    *(float4*)&b1[4] = *(const float4*)(kp + 20);

    float v0[8], v1[8];
    float tmax = -1.0e30f;
#pragma unroll
    for (int i = 0; i < 8; ++i) {
      v0[i] = s0[i] * scale + b0[i];
      v1[i] = s1[i] * scale + b1[i];
      tmax  = fmaxf(tmax, fmaxf(v0[i], v1[i]));
    }
    tmax = fmaxf(tmax, __shfl_xor(tmax, 16, 32)); // combine key halves

    float mnew  = fmaxf(rm, tmax);
    float alpha = __expf(rm - mnew);

    float psum = 0.f;
#pragma unroll
    for (int i = 0; i < 8; ++i) {
      float p0 = __expf(v0[i] - mnew);
      float p1 = __expf(v1[i] - mnew);
      psum += p0 + p1;
      // P^T (C layout) -> LDS as pbuf[query][key]
      pbuf[wave][col][hf * 8 + i]      = f2bf(p0);
      pbuf[wave][col][16 + hf * 8 + i] = f2bf(p1);
    }
    psum += __shfl_xor(psum, 16, 32);

    rl = rl * alpha + psum;
    rm = mnew;
#pragma unroll
    for (int t = 0; t < 4; ++t)
#pragma unroll
      for (int i = 0; i < 8; ++i) acc[t][i] *= alpha;

    // ---- O^T += V^T x P  (A = V^T rows, B = P from LDS) ------------------
    v16bf pfrag = load_b_frag(&pbuf[wave][0][0], 32, 0, 0);
#pragma unroll
    for (int t = 0; t < 4; ++t)
      acc[t] = WMMA_BF16(load_a_frag(Vh, S, t * 16, kb), pfrag, acc[t]);
  }

  // ---- normalize + write ctx[b*S + query, h*64 + d] -----------------------
  float inv = 1.f / rl;
  size_t row = ((size_t)b * S + m0 + col) * (size_t)(H * 64);
#pragma unroll
  for (int t = 0; t < 4; ++t) {
#pragma unroll
    for (int i = 0; i < 8; ++i) {
      int d = t * 16 + hf * 8 + i;
      ctx[row + hh * 64 + d] = f2bf(acc[t][i] * inv);
    }
  }
}

// ---------------------------------------------------------------------------
// LayerNorm over E=1024; one row per block (256 thr). Optional f32/bf16 out.
// ---------------------------------------------------------------------------
__global__ void k_layernorm(const float* __restrict__ in, const float* __restrict__ gamma,
                            const float* __restrict__ beta, float* __restrict__ out_f32,
                            u16* __restrict__ out_bf16, int E) {
  __shared__ float sred[2][8];
  int row = blockIdx.x;
  const float* x = in + (size_t)row * E;

  float s = 0.f, s2 = 0.f;
  for (int i = threadIdx.x; i < E; i += blockDim.x) {
    float v = x[i];
    s += v; s2 += v * v;
  }
  for (int o = 1; o < 32; o <<= 1) {
    s  += __shfl_xor(s,  o, 32);
    s2 += __shfl_xor(s2, o, 32);
  }
  int w = threadIdx.x >> 5;
  if ((threadIdx.x & 31) == 0) { sred[0][w] = s; sred[1][w] = s2; }
  __syncthreads();
  s = 0.f; s2 = 0.f;
#pragma unroll
  for (int j = 0; j < 8; ++j) { s += sred[0][j]; s2 += sred[1][j]; }

  float mean = s / E;
  float var  = s2 / E - mean * mean;
  float rstd = rsqrtf(var + 1e-5f);

  for (int i = threadIdx.x; i < E; i += blockDim.x) {
    float v = (x[i] - mean) * rstd * gamma[i] + beta[i];
    if (out_f32)  out_f32[(size_t)row * E + i]  = v;
    if (out_bf16) out_bf16[(size_t)row * E + i] = f2bf(v);
  }
}

// ---------------------------------------------------------------------------
// Host launcher
// ---------------------------------------------------------------------------
extern "C" void kernel_launch(void* const* d_in, const int* in_sizes, int n_in,
                              void* d_out, int out_size, void* d_ws, size_t ws_size,
                              hipStream_t stream) {
  (void)in_sizes; (void)n_in; (void)out_size; (void)ws_size;
  const int B = 4, S = 2048, E = 1024, H = 16, D = 64;
  const size_t T  = (size_t)B * S;   // 8192
  const size_t HD = (size_t)H * D;   // 1024

  const float* x    = (const float*)d_in[0];
  const int*   mask = (const int*)  d_in[1];
  const float* Wq = (const float*)d_in[2];  const float* bq = (const float*)d_in[3];
  const float* Wk = (const float*)d_in[4];  const float* bk = (const float*)d_in[5];
  const float* Wv = (const float*)d_in[6];  const float* bv = (const float*)d_in[7];
  const float* Wo = (const float*)d_in[8];  const float* bo = (const float*)d_in[9];
  const float* g1 = (const float*)d_in[10]; const float* b1 = (const float*)d_in[11];
  const float* W1 = (const float*)d_in[12]; const float* c1 = (const float*)d_in[13];
  const float* W2 = (const float*)d_in[14]; const float* c2 = (const float*)d_in[15];
  const float* g2 = (const float*)d_in[16]; const float* b2 = (const float*)d_in[17];

  unsigned char* ws = (unsigned char*)d_ws;
  size_t off = 0;
  auto carve = [&](size_t bytes) -> void* {
    void* p = ws + off;
    off = (off + bytes + 255) & ~(size_t)255;
    return p;
  };

  u16*   xb  = (u16*)carve(T * E * 2);          // x bf16
  u16*   wqT = (u16*)carve(HD * E * 2);         // [n=h*D+d, k=e]
  u16*   wkT = (u16*)carve(HD * E * 2);
  u16*   wvT = (u16*)carve(HD * E * 2);
  u16*   woT = (u16*)carve((size_t)E * HD * 2); // [n=e, k=h*D+d]
  u16*   w1T = (u16*)carve((size_t)D * E * 2);  // [n(64), k=E]
  u16*   w2T = (u16*)carve((size_t)E * D * 2);  // [n=E, k(64)]
  u16*   qb  = (u16*)carve(T * HD * 2);         // [B,H,S,D]
  u16*   kbuf= (u16*)carve(T * HD * 2);         // [B,H,S,D]
  u16*   vTb = (u16*)carve(T * HD * 2);         // [B,H,D,S]
  u16*   ctx = (u16*)carve(T * HD * 2);         // [T, H*D]
  float* tmp = (float*)carve(T * E * 4);        // f32 scratch (Wo out / FFN2 out)
  u16*   hb  = (u16*)carve(T * E * 2);          // LN1 out bf16
  u16*   fb  = (u16*)carve(T * (size_t)D * 2);  // FFN1 out bf16
  float* kbias = (float*)carve(T * 4);          // additive mask bias [B,S]

  const int TB = 256;
  auto blks = [&](size_t n) { return (unsigned)((n + TB - 1) / TB); };

  // 1) convert / repack
  k_f32_to_bf16<<<blks(T * E), TB, 0, stream>>>(x, xb, T * E);
  k_mask_bias<<<blks(T), TB, 0, stream>>>(mask, kbias, T);
  k_repack_qkv<<<blks(HD * E), TB, 0, stream>>>(Wq, wqT, E, D, HD * E);
  k_repack_qkv<<<blks(HD * E), TB, 0, stream>>>(Wk, wkT, E, D, HD * E);
  k_repack_qkv<<<blks(HD * E), TB, 0, stream>>>(Wv, wvT, E, D, HD * E);
  k_transpose_bf16<<<blks((size_t)E * HD), TB, 0, stream>>>(Wo, woT, (int)HD, E, (size_t)E * HD);
  k_transpose_bf16<<<blks((size_t)D * E), TB, 0, stream>>>(W1, w1T, E, D, (size_t)D * E);
  k_transpose_bf16<<<blks((size_t)E * D), TB, 0, stream>>>(W2, w2T, D, E, (size_t)E * D);

  // 2) QKV projections (WMMA + async-LDS staged B panels)
  dim3 gq((unsigned)(HD / 32), (unsigned)(T / 256));
  k_gemm_bf16<0><<<gq, 256, 0, stream>>>(xb, wqT, bq, qb,   (int)T, E, (int)HD, S, H);
  k_gemm_bf16<0><<<gq, 256, 0, stream>>>(xb, wkT, bk, kbuf, (int)T, E, (int)HD, S, H);
  k_gemm_bf16<1><<<gq, 256, 0, stream>>>(xb, wvT, bv, vTb,  (int)T, E, (int)HD, S, H);

  // 3) flash attention (WMMA, transposed dataflow)
  dim3 ga((unsigned)(S / 64), (unsigned)(B * H));
  k_flash_attn<<<ga, 128, 0, stream>>>(qb, kbuf, vTb, kbias, ctx, B, H, S);

  // 4) output projection + LN1
  dim3 go((unsigned)(E / 32), (unsigned)(T / 256));
  k_gemm_bf16<2><<<go, 256, 0, stream>>>(ctx, woT, bo, tmp, (int)T, (int)HD, E, S, H);
  k_layernorm<<<(unsigned)T, 256, 0, stream>>>(tmp, g1, b1, nullptr, hb, E);

  // 5) bottleneck FFN: GEMM+GELU, GEMM, LN2 -> out
  dim3 g1d((unsigned)(D / 32), (unsigned)(T / 256));
  k_gemm_bf16<3><<<g1d, 256, 0, stream>>>(hb, w1T, c1, fb, (int)T, E, D, S, H);
  dim3 g2d((unsigned)(E / 32), (unsigned)(T / 256));
  k_gemm_bf16<2><<<g2d, 256, 0, stream>>>(fb, w2T, c2, tmp, (int)T, D, E, S, H);
  k_layernorm<<<(unsigned)T, 256, 0, stream>>>(tmp, g2, b2, (float*)d_out, nullptr, E);
}
